// TransformerLanguageModel_28501402976276
// MI455X (gfx1250) — compile-verified, matrix operand order checked by reading, not run on hardware
//
#include <hip/hip_runtime.h>
#include <hip/hip_bf16.h>
#include <math.h>

typedef _Float16 f16;
typedef __attribute__((ext_vector_type(8)))  _Float16 v8h;
typedef __attribute__((ext_vector_type(16))) _Float16 v16h;
typedef __attribute__((ext_vector_type(8)))  float    v8f;

// Fixed model dims (from the reference harness)
constexpr int VOC = 32000, D = 1024, NH = 16, DH = 64, FF = 4096, NL = 6, NB = 2, TT = 1024;
constexpr int M = NB * TT;   // 2048 activation rows

// ---------------------------------------------------------------------------
// Fragment load helper: CDNA5 16-bit A/B fragment layout for 16x16x32 WMMA.
// lane l: row = l&15; halves 0..7 -> K = 8*(l>>4)+0..7 ; halves 8..15 -> K+16.
// Caller passes pointer to row start + 8*(l>>4) halves; chunks are 16B aligned.
// ---------------------------------------------------------------------------
__device__ __forceinline__ v16h ldfrag(const f16* p) {
  v8h lo = *(const v8h*)(p);
  v8h hi = *(const v8h*)(p + 16);
  v16h r;
#pragma unroll
  for (int i = 0; i < 8; ++i) { r[i] = lo[i]; r[i + 8] = hi[i]; }
  return r;
}

__device__ __forceinline__ v8f wmma16(v16h a, v16h b, v8f c) {
  return __builtin_amdgcn_wmma_f32_16x16x32_f16(false, a, false, b, (short)0, c,
                                                false, false);
}

// ---------------------------------------------------------------------------
// Embedding + interleaved sinusoidal positional encoding
// ---------------------------------------------------------------------------
__global__ void embed_kernel(const int* __restrict__ tok,
                             const float* __restrict__ emb,
                             float* __restrict__ x) {
  int idx = blockIdx.x * 256 + threadIdx.x;  // over M*D
  int row = idx >> 10;                       // D = 1024
  int d   = idx & (D - 1);
  int t   = row & (TT - 1);
  // ang = pos * exp(-(2p) * ln(10000)/D), dim 2p -> sin, 2p+1 -> cos
  float ang = (float)t * __expf(-(float)(d & ~1) * (9.21034037198f / (float)D));
  float pe  = (d & 1) ? __cosf(ang) : __sinf(ang);
  x[idx] = emb[(size_t)tok[row] * D + d] + pe;
}

// ---------------------------------------------------------------------------
// LayerNorm (block per row), emits f16 for the following WMMA GEMM
// ---------------------------------------------------------------------------
__global__ __launch_bounds__(256)
void ln_kernel(const float* __restrict__ x, const float* __restrict__ g,
               const float* __restrict__ b, f16* __restrict__ y) {
  const int row = blockIdx.x, tid = threadIdx.x;
  const float* xr = x + (size_t)row * D;
  float v[4], s = 0.f, s2 = 0.f;
#pragma unroll
  for (int i = 0; i < 4; ++i) {
    v[i] = xr[tid + i * 256];
    s += v[i]; s2 += v[i] * v[i];
  }
  __shared__ float r1[256], r2[256];
  r1[tid] = s; r2[tid] = s2;
  __syncthreads();
  for (int off = 128; off > 0; off >>= 1) {
    if (tid < off) { r1[tid] += r1[tid + off]; r2[tid] += r2[tid + off]; }
    __syncthreads();
  }
  float mean = r1[0] * (1.f / D);
  float var  = r2[0] * (1.f / D) - mean * mean;
  float rstd = rsqrtf(var + 1e-5f);
#pragma unroll
  for (int i = 0; i < 4; ++i) {
    int c = tid + i * 256;
    y[(size_t)row * D + c] = (f16)((v[i] - mean) * rstd * g[c] + b[c]);
  }
}

// ---------------------------------------------------------------------------
// Generic WMMA GEMM: C[M x ND] = A[M x KD] (f16) * Bw[KD x ND] (f32 weights,
// converted+transposed to f16 in LDS). 128x128 tile / workgroup; 8 waves in a
// 4(M) x 2(N) grid, each wave computes 32x64 via 2 A-frags x 4 B-frags.
// Double-buffered LDS with reg-staged global loads overlapping the WMMAs.
// ---------------------------------------------------------------------------
enum { EPI_QKV, EPI_RES, EPI_GELU, EPI_RESB, EPI_VOCAB };

template <int EPI, bool QKVB, int KD, int ND>
__global__ __launch_bounds__(256)
void gemm_kernel(const f16* __restrict__ A, const float* __restrict__ Bw,
                 const float* __restrict__ bias, float* __restrict__ xio,
                 float* __restrict__ outf, f16* __restrict__ out16) {
  __shared__ f16 sA[2][128][40];  // 128 rows x 32 K (pad 8)
  __shared__ f16 sB[2][128][40];  // B^T: 128 cols x 32 K (pad 8)
  const int tid  = threadIdx.x;
  const int lane = tid & 31, wv = tid >> 5;
  const int r = lane & 15, gi = lane >> 4;
  const int wm = wv & 3, wn = wv >> 2;
  const int tm = blockIdx.y * 128, tn = blockIdx.x * 128;

  v8f acc[2][4] = {};
  v8h aR[2];
  float4 bR[4];

  auto ldglobal = [&](int k0) {
#pragma unroll
    for (int it = 0; it < 2; ++it) {
      int idx = (it * 256 + tid) * 8;
      int row = idx >> 5, col = idx & 31;
      aR[it] = *(const v8h*)(A + (size_t)(tm + row) * KD + k0 + col);
    }
#pragma unroll
    for (int e = 0; e < 4; ++e) {
      int idx = (e * 256 + tid) * 4;
      int kk = idx >> 7, nn = idx & 127;
      int col = tn + nn, krow = k0 + kk;
      const float* src;
      if constexpr (QKVB)  // wq[l] layout [H, D, DH]; col j = h*DH + e
        src = Bw + ((size_t)(col >> 6) * KD + krow) * 64 + (col & 63);
      else
        src = Bw + (size_t)krow * ND + col;
      bR[e] = *(const float4*)src;
    }
  };
  auto stlds = [&](int buf) {
#pragma unroll
    for (int it = 0; it < 2; ++it) {
      int idx = (it * 256 + tid) * 8;
      int row = idx >> 5, col = idx & 31;
      *(v8h*)&sA[buf][row][col] = aR[it];
    }
#pragma unroll
    for (int e = 0; e < 4; ++e) {
      int idx = (e * 256 + tid) * 4;
      int kk = idx >> 7, nn = idx & 127;
      sB[buf][nn + 0][kk] = (f16)bR[e].x;
      sB[buf][nn + 1][kk] = (f16)bR[e].y;
      sB[buf][nn + 2][kk] = (f16)bR[e].z;
      sB[buf][nn + 3][kk] = (f16)bR[e].w;
    }
  };

  constexpr int NS = KD / 32;
  ldglobal(0);
  stlds(0);
  __syncthreads();
  for (int s = 0; s < NS; ++s) {
    const int cur = s & 1;
    if (s + 1 < NS) ldglobal((s + 1) * 32);
    if (s + 2 < NS) {  // speculative prefetch of the weight stream (L2)
      int k0n = (s + 2) * 32;
      int kk = tid >> 3, nn = (tid & 7) * 16;
      const float* pf;
      if constexpr (QKVB)
        pf = Bw + ((size_t)((tn + nn) >> 6) * KD + k0n + kk) * 64 + ((tn + nn) & 63);
      else
        pf = Bw + (size_t)(k0n + kk) * ND + tn + nn;
      __builtin_prefetch(pf, 0, 0);
    }
    v16h a0 = ldfrag(&sA[cur][wm * 32 + r][8 * gi]);
    v16h a1 = ldfrag(&sA[cur][wm * 32 + 16 + r][8 * gi]);
#pragma unroll
    for (int nt = 0; nt < 4; ++nt) {
      v16h bf = ldfrag(&sB[cur][wn * 64 + nt * 16 + r][8 * gi]);
      acc[0][nt] = wmma16(a0, bf, acc[0][nt]);
      acc[1][nt] = wmma16(a1, bf, acc[1][nt]);
    }
    if (s + 1 < NS) stlds(cur ^ 1);
    __syncthreads();
  }

  // Epilogue. C layout: lane l -> n = l&15; VGPR v -> m = v + 8*(l>>4)
#pragma unroll
  for (int ai = 0; ai < 2; ++ai) {
#pragma unroll
    for (int nt = 0; nt < 4; ++nt) {
#pragma unroll
      for (int vv = 0; vv < 8; ++vv) {
        float val = acc[ai][nt][vv];
        int row = tm + wm * 32 + ai * 16 + vv + 8 * gi;
        int col = tn + wn * 64 + nt * 16 + r;
        if constexpr (EPI == EPI_QKV) {
          // scatter into [B, H, T, DH]
          int bb = row >> 10, t = row & (TT - 1);
          int hh = col >> 6, e = col & 63;
          out16[(((size_t)bb * NH + hh) * TT + t) * DH + e] = (f16)val;
        } else if constexpr (EPI == EPI_RES) {
          size_t i = (size_t)row * D + col;
          xio[i] = xio[i] + val;
        } else if constexpr (EPI == EPI_GELU) {
          float z = val + bias[col];
          float gl = 0.5f * z * (1.f + erff(z * 0.70710678118654752f));
          out16[(size_t)row * FF + col] = (f16)gl;
        } else if constexpr (EPI == EPI_RESB) {
          size_t i = (size_t)row * D + col;
          xio[i] = xio[i] + val + bias[col];
        } else {  // EPI_VOCAB
          outf[(size_t)row * VOC + col] = val + bias[col];
        }
      }
    }
  }
}

// ---------------------------------------------------------------------------
// Flash attention: workgroup = (b, h, 128-query block); wave = 16-row Q tile.
// Streams 32-key blocks with online softmax; causal skipping per wave.
// ---------------------------------------------------------------------------
__global__ __launch_bounds__(256)
void attn_kernel(const f16* __restrict__ q, const f16* __restrict__ k,
                 const f16* __restrict__ v, f16* __restrict__ o16) {
  __shared__ f16 sK[32][72];       // K block row-major [key][dh] (== B^T for QK^T)
  __shared__ f16 sVt[64][40];      // V^T block [dh][key]
  __shared__ f16 sP[8][16][40];    // per-wave probability tile [m][key]
  const int tid  = threadIdx.x;
  const int lane = tid & 31, wv = tid >> 5;
  const int r = lane & 15, gi = lane >> 4;
  const int bh = blockIdx.y, qblk = blockIdx.x;
  const int bb = bh >> 4, hh = bh & 15;
  const size_t base = (size_t)bh * TT * DH;
  const int q0 = qblk * 128 + wv * 16;

  // Q fragments (K = dh 0..31 and 32..63), pre-scaled by 1/sqrt(DH)
  v16h qa0 = ldfrag(q + base + (size_t)(q0 + r) * DH + 8 * gi);
  v16h qa1 = ldfrag(q + base + (size_t)(q0 + r) * DH + 32 + 8 * gi);
#pragma unroll
  for (int i = 0; i < 16; ++i) {
    qa0[i] = (f16)((float)qa0[i] * 0.125f);
    qa1[i] = (f16)((float)qa1[i] * 0.125f);
  }

  float mrow[8], lrow[8];
  v8f oc[4] = {};
#pragma unroll
  for (int i = 0; i < 8; ++i) { mrow[i] = -1e30f; lrow[i] = 0.f; }

  const int nkb   = qblk * 4 + 4;        // key blocks any wave in block needs
  const int mykb  = (q0 + 15) / 32 + 1;  // key blocks THIS wave needs (causal)
  for (int kb = 0; kb < nkb; ++kb) {
    __syncthreads();
    {  // stage K block (2048 halves, vectorized)
      int idx = tid * 8;
      int row = idx >> 6, col = idx & 63;
      *(v8h*)&sK[row][col] =
          *(const v8h*)(k + base + (size_t)(kb * 32 + row) * DH + col);
    }
#pragma unroll
    for (int e = 0; e < 8; ++e) {  // stage V^T (coalesced read, transposed write)
      int idx = e * 256 + tid;
      int key = idx >> 6, dh = idx & 63;
      sVt[dh][key] = v[base + (size_t)(kb * 32 + key) * DH + dh];
    }
    __syncthreads();
    if (kb >= mykb) continue;  // fully-masked block for this wave

    // S = Q K^T for 2 n-tiles (keys 0..15 / 16..31), K-dim = 64 in two chunks
    v8f s0 = {}, s1 = {};
    {
      v16h b0 = ldfrag(&sK[r][8 * gi]);
      v16h b1 = ldfrag(&sK[r][32 + 8 * gi]);
      s0 = wmma16(qa0, b0, s0);
      s0 = wmma16(qa1, b1, s0);
      v16h b2 = ldfrag(&sK[16 + r][8 * gi]);
      v16h b3 = ldfrag(&sK[16 + r][32 + 8 * gi]);
      s1 = wmma16(qa0, b2, s1);
      s1 = wmma16(qa1, b3, s1);
    }

    // causal mask + online softmax (row reductions across 16-lane half groups)
    const int key0 = kb * 32 + r, key1 = key0 + 16;
#pragma unroll
    for (int vv = 0; vv < 8; ++vv) {
      int qi = q0 + vv + 8 * gi;
      float a0 = (key0 <= qi) ? s0[vv] : -1e30f;
      float a1 = (key1 <= qi) ? s1[vv] : -1e30f;
      float mx = fmaxf(a0, a1);
#pragma unroll
      for (int off = 8; off >= 1; off >>= 1)
        mx = fmaxf(mx, __shfl_xor(mx, off, 32));
      float nm = fmaxf(mrow[vv], mx);
      float alpha = __expf(mrow[vv] - nm);
      mrow[vv] = nm;
      float p0 = __expf(a0 - nm);
      float p1 = __expf(a1 - nm);
      float rs = p0 + p1;
#pragma unroll
      for (int off = 8; off >= 1; off >>= 1) rs += __shfl_xor(rs, off, 32);
      lrow[vv] = lrow[vv] * alpha + rs;
#pragma unroll
      for (int t = 0; t < 4; ++t) oc[t][vv] *= alpha;
      sP[wv][vv + 8 * gi][r]      = (f16)p0;
      sP[wv][vv + 8 * gi][16 + r] = (f16)p1;
    }
    // per-wave LDS round trip: LDS is in-order within a wave, just fence it
    __asm__ volatile("s_wait_dscnt 0" ::: "memory");

    // O += P V  (A = P[16x32], B^T = V^T rows per dh n-tile)
    v16h pa = ldfrag(&sP[wv][r][8 * gi]);
#pragma unroll
    for (int t = 0; t < 4; ++t) {
      v16h bv = ldfrag(&sVt[t * 16 + r][8 * gi]);
      oc[t] = wmma16(pa, bv, oc[t]);
    }
  }

  // normalize + scatter to concat-head layout [B, T, H*DH] (f16)
  float invl[8];
#pragma unroll
  for (int vv = 0; vv < 8; ++vv) invl[vv] = 1.f / lrow[vv];
#pragma unroll
  for (int t = 0; t < 4; ++t)
#pragma unroll
    for (int vv = 0; vv < 8; ++vv) {
      int row = q0 + vv + 8 * gi;
      o16[((size_t)bb * TT + row) * D + hh * DH + t * 16 + r] =
          (f16)(oc[t][vv] * invl[vv]);
    }
}

// ---------------------------------------------------------------------------
// Host orchestration
// ---------------------------------------------------------------------------
extern "C" void kernel_launch(void* const* d_in, const int* in_sizes, int n_in,
                              void* d_out, int out_size, void* d_ws,
                              size_t ws_size, hipStream_t stream) {
  const int*   tokens = (const int*)  d_in[0];
  const float* emb    = (const float*)d_in[1];
  const float* wq     = (const float*)d_in[2];
  const float* wk     = (const float*)d_in[3];
  const float* wv     = (const float*)d_in[4];
  const float* wo     = (const float*)d_in[5];
  const float* ln1_g  = (const float*)d_in[6];
  const float* ln1_b  = (const float*)d_in[7];
  const float* ln2_g  = (const float*)d_in[8];
  const float* ln2_b  = (const float*)d_in[9];
  const float* w1     = (const float*)d_in[10];
  const float* b1     = (const float*)d_in[11];
  const float* w2     = (const float*)d_in[12];
  const float* b2     = (const float*)d_in[13];
  const float* lnf_g  = (const float*)d_in[14];
  const float* lnf_b  = (const float*)d_in[15];
  const float* w_voc  = (const float*)d_in[16];
  const float* b_voc  = (const float*)d_in[17];
  float* out = (float*)d_out;

  char* ws = (char*)d_ws;
  float* x  = (float*)ws;               ws += (size_t)M * D * sizeof(float);
  f16* h16  = (f16*)ws;                 ws += (size_t)M * D * sizeof(f16);
  f16* q16  = (f16*)ws;                 ws += (size_t)M * D * sizeof(f16);
  f16* k16  = (f16*)ws;                 ws += (size_t)M * D * sizeof(f16);
  f16* v16  = (f16*)ws;                 ws += (size_t)M * D * sizeof(f16);
  f16* o16  = (f16*)ws;                 ws += (size_t)M * D * sizeof(f16);
  f16* g16  = (f16*)ws;                 ws += (size_t)M * FF * sizeof(f16);

  const dim3 blk(256);
  const dim3 gD (D  / 128, M / 128);   // N=1024 GEMMs
  const dim3 gF (FF / 128, M / 128);   // N=4096 GEMM
  const dim3 gV (VOC / 128, M / 128);  // N=32000 GEMM

  embed_kernel<<<M * D / 256, blk, 0, stream>>>(tokens, emb, x);

  for (int l = 0; l < NL; ++l) {
    const float* wql = wq + (size_t)l * NH * D * DH;
    const float* wkl = wk + (size_t)l * NH * D * DH;
    const float* wvl = wv + (size_t)l * NH * D * DH;

    ln_kernel<<<M, blk, 0, stream>>>(x, ln1_g + l * D, ln1_b + l * D, h16);

    gemm_kernel<EPI_QKV, true, D, D><<<gD, blk, 0, stream>>>(
        h16, wql, nullptr, nullptr, nullptr, q16);
    gemm_kernel<EPI_QKV, true, D, D><<<gD, blk, 0, stream>>>(
        h16, wkl, nullptr, nullptr, nullptr, k16);
    gemm_kernel<EPI_QKV, true, D, D><<<gD, blk, 0, stream>>>(
        h16, wvl, nullptr, nullptr, nullptr, v16);

    attn_kernel<<<dim3(TT / 128, NB * NH), blk, 0, stream>>>(q16, k16, v16, o16);

    gemm_kernel<EPI_RES, false, D, D><<<gD, blk, 0, stream>>>(
        o16, wo + (size_t)l * D * D, nullptr, x, nullptr, nullptr);

    ln_kernel<<<M, blk, 0, stream>>>(x, ln2_g + l * D, ln2_b + l * D, h16);

    gemm_kernel<EPI_GELU, false, D, FF><<<gF, blk, 0, stream>>>(
        h16, w1 + (size_t)l * D * FF, b1 + (size_t)l * FF, nullptr, nullptr, g16);

    gemm_kernel<EPI_RESB, false, FF, D><<<gD, blk, 0, stream>>>(
        g16, w2 + (size_t)l * FF * D, b2 + (size_t)l * D, x, nullptr, nullptr);
  }

  ln_kernel<<<M, blk, 0, stream>>>(x, lnf_g, lnf_b, h16);

  gemm_kernel<EPI_VOCAB, false, D, VOC><<<gV, blk, 0, stream>>>(
      h16, w_voc, b_voc, nullptr, out, nullptr);
}
